// Transformer_XL_55018531062082
// MI455X (gfx1250) — compile-verified, matrix operand order checked by reading, not run on hardware
//
#include <hip/hip_runtime.h>
#include <hip/hip_bf16.h>
#include <math.h>

// ---------------------------------------------------------------------------
// Fixed problem dimensions (from reference setup_inputs)
// ---------------------------------------------------------------------------
constexpr int Bc  = 2;
constexpr int Sc  = 1024;
constexpr int Mc  = 1024;
constexpr int Tc  = 2048;     // M + S
constexpr int Dc  = 1024;
constexpr int Hc  = 16;
constexpr int DHc = 64;       // D / H
constexpr int Lc  = 4;
constexpr int Vc  = 32000;

typedef __attribute__((ext_vector_type(16))) __bf16 v16bf;
typedef __attribute__((ext_vector_type(8)))  float  v8f;

union FragBF {
  v16bf          v;
  unsigned int   u[8];
  unsigned short h[16];
};

__device__ inline unsigned short f2bf(float f) {
  unsigned int u = __float_as_uint(f);
  u += 0x7FFFu + ((u >> 16) & 1u);   // round-to-nearest-even
  return (unsigned short)(u >> 16);
}

// ---------------------------------------------------------------------------
// gfx1250 async global->LDS copy (ASYNCcnt-tracked), 16B per lane.
// ---------------------------------------------------------------------------
__device__ inline void async_copy_b128(const void* gptr, void* ldsptr) {
  asm volatile("global_load_async_to_lds_b128 %0, %1, off"
               :: "v"((unsigned)(unsigned long long)ldsptr),
                  "v"((unsigned long long)gptr)
               : "memory");
}
__device__ inline void async_wait0() {
#if __has_builtin(__builtin_amdgcn_s_wait_asynccnt)
  __builtin_amdgcn_s_wait_asynccnt(0);
#else
  asm volatile("s_wait_asynccnt 0x0" ::: "memory");
#endif
}

// ---------------------------------------------------------------------------
// Tiled bf16 GEMM with f32 accumulation via v_wmma_f32_16x16x32_bf16.
//   C[z] = A[z] (Mdim x Kdim, row-major bf16)
//        @ (TRANSB ? B[z%bmod]^T : B[z%bmod]) + bias
//   !TRANSB: B row-major Kdim x Ndim.  TRANSB: B row-major Ndim x Kdim.
// 256 threads = 8 wave32; macro-tile 128(M) x BN(N); K step 32.
// Waves arranged 4(M) x 2(N); each wave owns 32 x BN/2 = 2 x SN WMMA subtiles.
// B tile is kept TRANSPOSED in LDS ([BN][BK]) so both operand fragments are
// contiguous 16B reads (-> ds_load_b128).  A tile and NT B tile are staged
// with async global->LDS b128 copies; NN B tile is transposed during staging.
// ---------------------------------------------------------------------------
template <bool TRANSB, int BN>
__global__ __launch_bounds__(256) void gemm_bf16_kernel(
    const unsigned short* __restrict__ Aall,
    const unsigned short* __restrict__ Ball,
    float* __restrict__ Call,
    const float* __restrict__ bias,
    int Mdim, int Ndim, int Kdim,
    long long sA, long long sB, long long sC, int bmod) {
  constexpr int BM = 128, BK = 32;
  constexpr int SN = BN / 32;             // N-subtiles per wave (wave spans BN/2)
  __shared__ unsigned short As[BM][BK];
  __shared__ unsigned short Bs[BN][BK];   // transposed tile: [n][k]

  const int z = blockIdx.z;
  const unsigned short* A  = Aall + (size_t)z * (size_t)sA;
  const unsigned short* Bm = Ball + (size_t)(z % bmod) * (size_t)sB;
  float* C = Call + (size_t)z * (size_t)sC;

  const int tid  = threadIdx.x;
  const int lane = tid & 31;
  const int wave = tid >> 5;
  const int wm   = wave >> 1;             // 0..3
  const int wn   = wave & 1;              // 0..1
  const int m0   = blockIdx.y * BM;
  const int n0   = blockIdx.x * BN;
  const int half = lane >> 4;
  const int ln   = lane & 15;

  v8f acc[2][SN] = {};

  for (int k0 = 0; k0 < Kdim; k0 += BK) {
    // ---- stage A tile (BM x BK), async b128: 4 chunks/row, 512 chunks ----
    for (int it = 0; it < (BM * BK / 8) / 256; ++it) {
      int chunk = tid + 256 * it;
      int r = chunk >> 2;                 // BK/8 = 4 chunks per row
      int c = (chunk & 3) * 8;
      async_copy_b128(&A[(size_t)(m0 + r) * Kdim + k0 + c], &As[r][c]);
    }
    // ---- stage B tile ----
    if (TRANSB) {
      // B row-major (N x K): rows map 1:1 to Bs rows -> async b128 copy.
      for (int it = 0; it < (BN * BK / 8) / 256; ++it) {
        int chunk = tid + 256 * it;
        int r = chunk >> 2;
        int c = (chunk & 3) * 8;
        async_copy_b128(&Bm[(size_t)(n0 + r) * Kdim + k0 + c], &Bs[r][c]);
      }
    } else {
      // B row-major (K x N): coalesced uint loads, transpose into Bs[n][k].
      for (int idx = tid; idx < BK * (BN / 2); idx += 256) {
        int r = idx / (BN / 2);           // k within tile
        int c = (idx % (BN / 2)) * 2;     // n pair within tile
        unsigned int w =
            *(const unsigned int*)&Bm[(size_t)(k0 + r) * Ndim + n0 + c];
        Bs[c][r]     = (unsigned short)w;
        Bs[c + 1][r] = (unsigned short)(w >> 16);
      }
    }
    async_wait0();
    __syncthreads();

    // ---- per-wave fragments (ISA 7.12.2 16-bit layout; contiguous 16B) ----
    FragBF a[2], b[SN];
    for (int sm = 0; sm < 2; ++sm) {
      int m = wm * 32 + sm * 16 + ln;
      for (int r = 0; r < 8; ++r) {
        int kb = (r < 4) ? (8 * half + 2 * r) : (16 + 8 * half + 2 * (r - 4));
        a[sm].u[r] = *(const unsigned int*)&As[m][kb];
      }
    }
    for (int sn = 0; sn < SN; ++sn) {
      int n = wn * (BN / 2) + sn * 16 + ln;
      for (int r = 0; r < 8; ++r) {
        int kb = (r < 4) ? (8 * half + 2 * r) : (16 + 8 * half + 2 * (r - 4));
        b[sn].u[r] = *(const unsigned int*)&Bs[n][kb];
      }
    }

    for (int sm = 0; sm < 2; ++sm)
      for (int sn = 0; sn < SN; ++sn)
        acc[sm][sn] = __builtin_amdgcn_wmma_f32_16x16x32_bf16(
            false, a[sm].v, false, b[sn].v, (short)0, acc[sm][sn], false, false);
    __syncthreads();
  }

  // ---- store D (VGPR v: lanes 0-15 -> M=v, lanes 16-31 -> M=8+v) ----
  for (int sm = 0; sm < 2; ++sm) {
    for (int sn = 0; sn < SN; ++sn) {
      int col = n0 + wn * (BN / 2) + sn * 16 + ln;
      float bv = bias ? bias[col] : 0.0f;
      for (int v = 0; v < 8; ++v) {
        int row = m0 + wm * 32 + sm * 16 + half * 8 + v;
        C[(size_t)row * Ndim + col] = acc[sm][sn][v] + bv;
      }
    }
  }
}

// ---------------------------------------------------------------------------
// Rel-shift + mask + softmax.  One workgroup per score row (b,h,i).
// shift(Bhat)[i,j] = Bhat[i, j + (S-1) - i] inside the unmasked region.
// ---------------------------------------------------------------------------
__global__ __launch_bounds__(256) void score_softmax_kernel(
    const float* __restrict__ A, const float* __restrict__ Bh,
    unsigned short* __restrict__ P) {
  const int row = blockIdx.x;             // ((b*H + h)*S + i)
  const int i   = row & (Sc - 1);
  const float scale = 0.125f;             // 1/sqrt(64)
  __shared__ float red[256];
  const size_t base = (size_t)row * Tc;

  float v[8];
  float mx = -3.0e38f;
  for (int c = 0; c < 8; ++c) {
    int j = threadIdx.x + c * 256;
    float sc;
    if (j > i + Mc) {
      sc = -1.0e9f;                       // masked (reference adds -1e9)
    } else {
      sc = (A[base + j] + Bh[base + j + (Sc - 1) - i]) * scale;
    }
    v[c] = sc;
    mx = fmaxf(mx, sc);
  }
  red[threadIdx.x] = mx;
  __syncthreads();
  for (int s = 128; s > 0; s >>= 1) {
    if (threadIdx.x < s) red[threadIdx.x] = fmaxf(red[threadIdx.x], red[threadIdx.x + s]);
    __syncthreads();
  }
  mx = red[0];
  __syncthreads();

  float sum = 0.0f;
  for (int c = 0; c < 8; ++c) {
    v[c] = __expf(v[c] - mx);
    sum += v[c];
  }
  red[threadIdx.x] = sum;
  __syncthreads();
  for (int s = 128; s > 0; s >>= 1) {
    if (threadIdx.x < s) red[threadIdx.x] += red[threadIdx.x + s];
    __syncthreads();
  }
  float inv = 1.0f / red[0];
  for (int c = 0; c < 8; ++c) {
    int j = threadIdx.x + c * 256;
    P[base + j] = f2bf(v[c] * inv);
  }
}

// ---------------------------------------------------------------------------
// Residual add + LayerNorm.  One workgroup per row of D=1024.
// ---------------------------------------------------------------------------
__global__ __launch_bounds__(256) void add_ln_kernel(
    const float* __restrict__ X, const float* __restrict__ R,
    const float* __restrict__ g, const float* __restrict__ b,
    float* __restrict__ outF, unsigned short* __restrict__ outB) {
  const int row = blockIdx.x;
  __shared__ float red[256];
  const size_t base = (size_t)row * Dc;

  float x[4];
  float s = 0.0f;
  for (int c = 0; c < 4; ++c) {
    int d = threadIdx.x + c * 256;
    x[c] = X[base + d] + (R ? R[base + d] : 0.0f);
    s += x[c];
  }
  red[threadIdx.x] = s;
  __syncthreads();
  for (int t = 128; t > 0; t >>= 1) {
    if (threadIdx.x < t) red[threadIdx.x] += red[threadIdx.x + t];
    __syncthreads();
  }
  float mu = red[0] * (1.0f / Dc);
  __syncthreads();

  float vs = 0.0f;
  for (int c = 0; c < 4; ++c) {
    float t = x[c] - mu;
    vs += t * t;
  }
  red[threadIdx.x] = vs;
  __syncthreads();
  for (int t = 128; t > 0; t >>= 1) {
    if (threadIdx.x < t) red[threadIdx.x] += red[threadIdx.x + t];
    __syncthreads();
  }
  float rs = rsqrtf(red[0] * (1.0f / Dc) + 1e-5f);

  for (int c = 0; c < 4; ++c) {
    int d = threadIdx.x + c * 256;
    float y = (x[c] - mu) * rs * g[d] + b[d];
    if (outF) outF[base + d] = y;
    if (outB) outB[base + d] = f2bf(y);
  }
}

// ---------------------------------------------------------------------------
// Elementwise kernels (grid-stride)
// ---------------------------------------------------------------------------
__global__ void conv_bf16_kernel(const float* __restrict__ in,
                                 unsigned short* __restrict__ out, size_t n) {
  for (size_t i = (size_t)blockIdx.x * blockDim.x + threadIdx.x; i < n;
       i += (size_t)gridDim.x * blockDim.x)
    out[i] = f2bf(in[i]);
}

__global__ void embed_kernel(const int* __restrict__ ids,
                             const float* __restrict__ emb,
                             float* __restrict__ xf,
                             unsigned short* __restrict__ xb) {
  const size_t n = (size_t)Bc * Sc * Dc;
  for (size_t i = (size_t)blockIdx.x * blockDim.x + threadIdx.x; i < n;
       i += (size_t)gridDim.x * blockDim.x) {
    int tok = ids[i / Dc];
    float v = emb[(size_t)tok * Dc + (i % Dc)] * 32.0f;   // sqrt(D)=32
    xf[i] = v;
    xb[i] = f2bf(v);
  }
}

__global__ void relenc_kernel(unsigned short* __restrict__ out) {
  const size_t n = (size_t)Tc * Dc;
  for (size_t i = (size_t)blockIdx.x * blockDim.x + threadIdx.x; i < n;
       i += (size_t)gridDim.x * blockDim.x) {
    int t = (int)(i / Dc), d = (int)(i % Dc);
    float p = (float)(Tc - 1 - t);                         // reversed
    float ang = p * powf(10000.0f, -(float)(2 * (d >> 1)) / (float)Dc);
    float v = (d & 1) ? cosf(ang) : sinf(ang);
    out[i] = f2bf(v);
  }
}

__global__ void concat_kernel(const float* __restrict__ mem_l,
                              const float* __restrict__ xf,
                              unsigned short* __restrict__ xt) {
  const size_t n = (size_t)Bc * Tc * Dc;
  for (size_t i = (size_t)blockIdx.x * blockDim.x + threadIdx.x; i < n;
       i += (size_t)gridDim.x * blockDim.x) {
    size_t b = i / ((size_t)Tc * Dc);
    size_t rem = i % ((size_t)Tc * Dc);
    int t = (int)(rem / Dc), d = (int)(rem % Dc);
    float v = (t < Mc) ? mem_l[(b * Mc + t) * Dc + d]
                       : xf[(b * Sc + (t - Mc)) * Dc + d];
    xt[i] = f2bf(v);
  }
}

// (B, rows, H, dh) f32 -> (B, H, rows, dh) bf16
__global__ void split_heads_kernel(const float* __restrict__ in,
                                   unsigned short* __restrict__ out, int rows) {
  const size_t n = (size_t)Bc * rows * Dc;
  for (size_t i = (size_t)blockIdx.x * blockDim.x + threadIdx.x; i < n;
       i += (size_t)gridDim.x * blockDim.x) {
    int dd = (int)(i % DHc);
    int s  = (int)((i / DHc) % rows);
    int h  = (int)((i / ((size_t)DHc * rows)) % Hc);
    int b  = (int)(i / ((size_t)DHc * rows * Hc));
    out[i] = f2bf(in[((size_t)b * rows + s) * Dc + h * DHc + dd]);
  }
}

// (T, D) f32 -> (H, T, dh) bf16
__global__ void split_qr_kernel(const float* __restrict__ in,
                                unsigned short* __restrict__ out) {
  const size_t n = (size_t)Hc * Tc * DHc;
  for (size_t i = (size_t)blockIdx.x * blockDim.x + threadIdx.x; i < n;
       i += (size_t)gridDim.x * blockDim.x) {
    int dd = (int)(i % DHc);
    int t  = (int)((i / DHc) % Tc);
    int h  = (int)(i / ((size_t)DHc * Tc));
    out[i] = f2bf(in[(size_t)t * Dc + h * DHc + dd]);
  }
}

// (B, H, S, dh) f32 -> (B, S, D) bf16
__global__ void merge_heads_kernel(const float* __restrict__ in,
                                   unsigned short* __restrict__ out) {
  const size_t n = (size_t)Bc * Sc * Dc;
  for (size_t i = (size_t)blockIdx.x * blockDim.x + threadIdx.x; i < n;
       i += (size_t)gridDim.x * blockDim.x) {
    int d = (int)(i % Dc);
    int s = (int)((i / Dc) % Sc);
    int b = (int)(i / ((size_t)Dc * Sc));
    int h = d / DHc, dd = d % DHc;
    out[i] = f2bf(in[(((size_t)b * Hc + h) * Sc + s) * DHc + dd]);
  }
}

__global__ void gelu_bf16_kernel(const float* __restrict__ in,
                                 unsigned short* __restrict__ out, size_t n) {
  for (size_t i = (size_t)blockIdx.x * blockDim.x + threadIdx.x; i < n;
       i += (size_t)gridDim.x * blockDim.x) {
    float x = in[i];
    float g = 0.5f * x * (1.0f + erff(x * 0.70710678118654752f));  // exact GELU
    out[i] = f2bf(g);
  }
}

// ---------------------------------------------------------------------------
// Host orchestration
// ---------------------------------------------------------------------------
extern "C" void kernel_launch(void* const* d_in, const int* in_sizes, int n_in,
                              void* d_out, int out_size, void* d_ws, size_t ws_size,
                              hipStream_t stream) {
  (void)in_sizes; (void)n_in; (void)out_size; (void)ws_size;

  const int*   inputs = (const int*)d_in[0];
  const float* mems   = (const float*)d_in[1];
  const float* emb    = (const float*)d_in[2];
  const float* Wq     = (const float*)d_in[3];
  const float* bq     = (const float*)d_in[4];
  const float* Wke    = (const float*)d_in[5];
  const float* bke    = (const float*)d_in[6];
  const float* Wkr    = (const float*)d_in[7];
  const float* bkr    = (const float*)d_in[8];
  const float* Wv     = (const float*)d_in[9];
  const float* bv     = (const float*)d_in[10];
  const float* Wo     = (const float*)d_in[11];
  const float* bo     = (const float*)d_in[12];
  const float* ln1_g  = (const float*)d_in[13];
  const float* ln1_b  = (const float*)d_in[14];
  const float* W1     = (const float*)d_in[15];
  const float* b1     = (const float*)d_in[16];
  const float* W2     = (const float*)d_in[17];
  const float* b2     = (const float*)d_in[18];
  const float* ln2_g  = (const float*)d_in[19];
  const float* ln2_b  = (const float*)d_in[20];
  const float* Wh     = (const float*)d_in[21];
  const float* bh     = (const float*)d_in[22];
  const float* Wf     = (const float*)d_in[23];
  const float* bfb    = (const float*)d_in[24];
  float* logits = (float*)d_out;

  // ---- workspace carving ----
  char* ws = (char*)d_ws;
  size_t off = 0;
  auto alloc = [&](size_t bytes) -> void* {
    void* p = ws + off;
    off += (bytes + 255) & ~(size_t)255;
    return p;
  };
  const size_t DD = (size_t)Dc * Dc;

  unsigned short* WqB  = (unsigned short*)alloc((size_t)Lc * DD * 2);
  unsigned short* WkeB = (unsigned short*)alloc((size_t)Lc * DD * 2);
  unsigned short* WkrB = (unsigned short*)alloc((size_t)Lc * DD * 2);
  unsigned short* WvB  = (unsigned short*)alloc((size_t)Lc * DD * 2);
  unsigned short* WoB  = (unsigned short*)alloc((size_t)Lc * DD * 2);
  unsigned short* W1B  = (unsigned short*)alloc((size_t)Lc * DD * 2);
  unsigned short* W2B  = (unsigned short*)alloc((size_t)Lc * DD * 2);
  unsigned short* WhB  = (unsigned short*)alloc(DD * 2);
  unsigned short* WfB  = (unsigned short*)alloc((size_t)Dc * Vc * 2);
  unsigned short* relB = (unsigned short*)alloc((size_t)Tc * Dc * 2);

  float*          x_f   = (float*)alloc((size_t)Bc * Sc * Dc * 4);
  unsigned short* x_b   = (unsigned short*)alloc((size_t)Bc * Sc * Dc * 2);
  unsigned short* xt_b  = (unsigned short*)alloc((size_t)Bc * Tc * Dc * 2);
  float*          prj_f = (float*)alloc((size_t)Bc * Tc * Dc * 4);   // reused scratch
  unsigned short* q_b   = (unsigned short*)alloc((size_t)Bc * Sc * Dc * 2);
  unsigned short* k_b   = (unsigned short*)alloc((size_t)Bc * Tc * Dc * 2);
  unsigned short* v_b   = (unsigned short*)alloc((size_t)Bc * Tc * Dc * 2);
  unsigned short* qr_b  = (unsigned short*)alloc((size_t)Hc * Tc * DHc * 2);
  float*          A_f   = (float*)alloc((size_t)Bc * Hc * Sc * Tc * 4);
  float*          Bh_f  = (float*)alloc((size_t)Bc * Hc * Sc * Tc * 4);
  unsigned short* at_b  = (unsigned short*)alloc((size_t)Bc * Hc * Sc * Tc * 2);
  float*          ctx_f = (float*)alloc((size_t)Bc * Hc * Sc * DHc * 4);
  unsigned short* ctx_b = (unsigned short*)alloc((size_t)Bc * Sc * Dc * 2);
  float*          ao_f  = (float*)alloc((size_t)Bc * Sc * Dc * 4);
  float*          h1_f  = (float*)alloc((size_t)Bc * Sc * Dc * 4);
  unsigned short* h1_b  = (unsigned short*)alloc((size_t)Bc * Sc * Dc * 2);
  unsigned short* gl_b  = (unsigned short*)alloc((size_t)Bc * Sc * Dc * 2);
  float*          f2_f  = (float*)alloc((size_t)Bc * Sc * Dc * 4);
  unsigned short* g_b   = (unsigned short*)alloc((size_t)Bc * Sc * Dc * 2);

  auto ew = [&](size_t n) -> dim3 {
    size_t blk = (n + 1023) / 1024;
    if (blk > 65535u * 16u) blk = 65535u * 16u;
    return dim3((unsigned)blk);
  };
  auto conv = [&](const float* in, unsigned short* out, size_t n) {
    conv_bf16_kernel<<<ew(n), 256, 0, stream>>>(in, out, n);
  };
  // BN=128 wide-tile GEMM (N % 128 == 0)
  auto gemm128 = [&](bool transB, const unsigned short* A, const unsigned short* B,
                     float* C, const float* bias, int Md, int Nd, int Kd,
                     long long sA, long long sB, long long sC, int batch, int bmod) {
    dim3 grid(Nd / 128, Md / 128, batch);
    if (transB)
      gemm_bf16_kernel<true, 128><<<grid, 256, 0, stream>>>(A, B, C, bias, Md, Nd, Kd, sA, sB, sC, bmod);
    else
      gemm_bf16_kernel<false, 128><<<grid, 256, 0, stream>>>(A, B, C, bias, Md, Nd, Kd, sA, sB, sC, bmod);
  };
  // BN=64 GEMM (for N = dh = 64)
  auto gemm64nn = [&](const unsigned short* A, const unsigned short* B,
                      float* C, int Md, int Nd, int Kd,
                      long long sA, long long sB, long long sC, int batch, int bmod) {
    dim3 grid(Nd / 64, Md / 128, batch);
    gemm_bf16_kernel<false, 64><<<grid, 256, 0, stream>>>(A, B, C, nullptr, Md, Nd, Kd, sA, sB, sC, bmod);
  };

  // ---- weight conversion to bf16 (every call; determinism) ----
  conv(Wq,  WqB,  (size_t)Lc * DD);
  conv(Wke, WkeB, (size_t)Lc * DD);
  conv(Wkr, WkrB, (size_t)Lc * DD);
  conv(Wv,  WvB,  (size_t)Lc * DD);
  conv(Wo,  WoB,  (size_t)Lc * DD);
  conv(W1,  W1B,  (size_t)Lc * DD);
  conv(W2,  W2B,  (size_t)Lc * DD);
  conv(Wh,  WhB,  DD);
  conv(Wf,  WfB,  (size_t)Dc * Vc);

  relenc_kernel<<<ew((size_t)Tc * Dc), 256, 0, stream>>>(relB);
  embed_kernel<<<ew((size_t)Bc * Sc * Dc), 256, 0, stream>>>(inputs, emb, x_f, x_b);

  const long long sQ = (long long)Sc * DHc;    // per-(b,h) q stride
  const long long sK = (long long)Tc * DHc;    // per-(b,h) k/v/qr stride
  const long long sS = (long long)Sc * Tc;     // per-(b,h) score stride

  for (int l = 0; l < Lc; ++l) {
    const float* bq_l  = bq  + (size_t)l * Dc;
    const float* bke_l = bke + (size_t)l * Dc;
    const float* bkr_l = bkr + (size_t)l * Dc;
    const float* bv_l  = bv  + (size_t)l * Dc;
    const float* bo_l  = bo  + (size_t)l * Dc;
    const float* b1_l  = b1  + (size_t)l * Dc;
    const float* b2_l  = b2  + (size_t)l * Dc;

    // x_tilde = concat(mems[l], x)  -> bf16
    concat_kernel<<<ew((size_t)Bc * Tc * Dc), 256, 0, stream>>>(
        mems + (size_t)l * Bc * Mc * Dc, x_f, xt_b);

    // Q = x @ Wq + bq ; split heads
    gemm128(false, x_b, WqB + (size_t)l * DD, prj_f, bq_l, Bc * Sc, Dc, Dc, 0, 0, 0, 1, 1);
    split_heads_kernel<<<ew((size_t)Bc * Sc * Dc), 256, 0, stream>>>(prj_f, q_b, Sc);

    // K = x_tilde @ Wke + bke ; split heads
    gemm128(false, xt_b, WkeB + (size_t)l * DD, prj_f, bke_l, Bc * Tc, Dc, Dc, 0, 0, 0, 1, 1);
    split_heads_kernel<<<ew((size_t)Bc * Tc * Dc), 256, 0, stream>>>(prj_f, k_b, Tc);

    // V = x_tilde @ Wv + bv ; split heads
    gemm128(false, xt_b, WvB + (size_t)l * DD, prj_f, bv_l, Bc * Tc, Dc, Dc, 0, 0, 0, 1, 1);
    split_heads_kernel<<<ew((size_t)Bc * Tc * Dc), 256, 0, stream>>>(prj_f, v_b, Tc);

    // Qr = rel_enc @ Wkr + bkr ; split to (H,T,dh)
    gemm128(false, relB, WkrB + (size_t)l * DD, prj_f, bkr_l, Tc, Dc, Dc, 0, 0, 0, 1, 1);
    split_qr_kernel<<<ew((size_t)Hc * Tc * DHc), 256, 0, stream>>>(prj_f, qr_b);

    // A = Q @ K^T       (batched over B*H)
    gemm128(true, q_b, k_b, A_f, nullptr, Sc, Tc, DHc, sQ, sK, sS, Bc * Hc, Bc * Hc);
    // Bhat = Q @ Qr^T   (B operand batched over H only)
    gemm128(true, q_b, qr_b, Bh_f, nullptr, Sc, Tc, DHc, sQ, sK, sS, Bc * Hc, Hc);

    // rel-shift + mask + softmax -> bf16 probabilities
    score_softmax_kernel<<<dim3(Bc * Hc * Sc), 256, 0, stream>>>(A_f, Bh_f, at_b);

    // ctx = P @ V (batched, N = dh = 64) ; merge heads -> bf16
    gemm64nn(at_b, v_b, ctx_f, Sc, DHc, Tc, sS, sK, sQ, Bc * Hc, Bc * Hc);
    merge_heads_kernel<<<ew((size_t)Bc * Sc * Dc), 256, 0, stream>>>(ctx_f, ctx_b);

    // attn_out = ctx @ Wo + bo
    gemm128(false, ctx_b, WoB + (size_t)l * DD, ao_f, bo_l, Bc * Sc, Dc, Dc, 0, 0, 0, 1, 1);

    // h1 = LN(x + attn_out)
    add_ln_kernel<<<dim3(Bc * Sc), 256, 0, stream>>>(
        x_f, ao_f, ln1_g + (size_t)l * Dc, ln1_b + (size_t)l * Dc, h1_f, h1_b);

    // ff = gelu(h1 @ W1 + b1) @ W2 + b2
    gemm128(false, h1_b, W1B + (size_t)l * DD, prj_f, b1_l, Bc * Sc, Dc, Dc, 0, 0, 0, 1, 1);
    gelu_bf16_kernel<<<ew((size_t)Bc * Sc * Dc), 256, 0, stream>>>(
        prj_f, gl_b, (size_t)Bc * Sc * Dc);
    gemm128(false, gl_b, W2B + (size_t)l * DD, f2_f, b2_l, Bc * Sc, Dc, Dc, 0, 0, 0, 1, 1);

    // x = LN(h1 + ff)
    add_ln_kernel<<<dim3(Bc * Sc), 256, 0, stream>>>(
        h1_f, f2_f, ln2_g + (size_t)l * Dc, ln2_b + (size_t)l * Dc, x_f, x_b);
  }

  // head: g = gelu(x @ Wh + bh) ; logits = g @ Wf + bf
  gemm128(false, x_b, WhB, prj_f, bh, Bc * Sc, Dc, Dc, 0, 0, 0, 1, 1);
  gelu_bf16_kernel<<<ew((size_t)Bc * Sc * Dc), 256, 0, stream>>>(
      prj_f, g_b, (size_t)Bc * Sc * Dc);
  gemm128(false, g_b, WfB, logits, bfb, Bc * Sc, Vc, Dc, 0, 0, 0, 1, 1);
}
